// Net_48911087567423
// MI455X (gfx1250) — 8-bit matrix operand order repaired and checked by reading, compiled, not run on hardware
//
#include <hip/hip_runtime.h>

typedef int v8i __attribute__((ext_vector_type(8)));

#define T_STEPS 25
#define BATCH   4096
#define NIN     784
#define NH      1000
#define NOUT    10
#define NH_PAD  1024
#define K1_PAD  832
#define K2      1024
#define THRV    0.1f

#define OFF_MAX    0
#define OFF_SCALE  16
#define OFF_W1Q    256
#define OFF_W2Q    (OFF_W1Q + NH_PAD * K1_PAD)
#define OFF_SPK1   (OFF_W2Q + 16 * K2)
#define OFF_MEM1   (OFF_SPK1 + BATCH * NH_PAD)
#define OFF_MEM2   (OFF_MEM1 + BATCH * NH_PAD * 4)
#define WS_END     (OFF_MEM2 + BATCH * 16 * 4)

__device__ __forceinline__ void async_copy64(unsigned lds_off,
                                             const void* gptr) {
    const unsigned long long ga = (unsigned long long)(size_t)gptr;
    asm volatile(
        "global_load_async_to_lds_b128 %0, %1, off\n\t"
        "global_load_async_to_lds_b128 %0, %1, off offset:16\n\t"
        "global_load_async_to_lds_b128 %0, %1, off offset:32\n\t"
        "global_load_async_to_lds_b128 %0, %1, off offset:48"
        :: "v"(lds_off), "v"(ga) : "memory");
}
__device__ __forceinline__ void async_wait0() {
    asm volatile("s_wait_asynccnt 0x0" ::: "memory");
}

__global__ void snn_init(unsigned char* __restrict__ ws) {
    const size_t stride = (size_t)gridDim.x * blockDim.x;
    size_t i = (size_t)blockIdx.x * blockDim.x + threadIdx.x;
    unsigned int* z = (unsigned int*)(ws + OFF_SPK1);
    const size_t nw = (size_t)(WS_END - OFF_SPK1) / 4;
    for (size_t j = i; j < nw; j += stride) z[j] = 0u;
    if (i < 2) ((unsigned int*)(ws + OFF_MAX))[i] = 0u;
}

__global__ void snn_absmax(const float* __restrict__ W1, const float* __restrict__ W2,
                           unsigned char* __restrict__ ws) {
    __shared__ unsigned int sm1, sm2;
    if (threadIdx.x == 0) { sm1 = 0u; sm2 = 0u; }
    __syncthreads();
    const size_t stride = (size_t)gridDim.x * blockDim.x;
    size_t i = (size_t)blockIdx.x * blockDim.x + threadIdx.x;
    float m1 = 0.f, m2 = 0.f;
    const size_t n1 = (size_t)NH * NIN, n2 = (size_t)NOUT * NH;
    for (size_t j = i; j < n1; j += stride) m1 = fmaxf(m1, fabsf(W1[j]));
    for (size_t j = i; j < n2; j += stride) m2 = fmaxf(m2, fabsf(W2[j]));
    atomicMax(&sm1, __float_as_uint(m1));
    atomicMax(&sm2, __float_as_uint(m2));
    __syncthreads();
    if (threadIdx.x == 0) {
        atomicMax((unsigned int*)(ws + OFF_MAX) + 0, sm1);
        atomicMax((unsigned int*)(ws + OFF_MAX) + 1, sm2);
    }
}

__global__ void snn_quant(const float* __restrict__ W1, const float* __restrict__ W2,
                          unsigned char* __restrict__ ws) {
    const unsigned int* mx = (const unsigned int*)(ws + OFF_MAX);
    float s1 = __uint_as_float(mx[0]) / 127.f;
    float s2 = __uint_as_float(mx[1]) / 127.f;
    if (!(s1 > 0.f)) s1 = 1.f;
    if (!(s2 > 0.f)) s2 = 1.f;
    signed char* W1q = (signed char*)(ws + OFF_W1Q);
    signed char* W2q = (signed char*)(ws + OFF_W2Q);
    const size_t N1 = (size_t)NH_PAD * K1_PAD;
    const size_t N2 = (size_t)16 * K2;
    const size_t stride = (size_t)gridDim.x * blockDim.x;
    size_t i = (size_t)blockIdx.x * blockDim.x + threadIdx.x;
    for (size_t j = i; j < N1 + N2; j += stride) {
        if (j < N1) {
            const int n = (int)(j / K1_PAD), k = (int)(j % K1_PAD);
            signed char q = 0;
            if (n < NH && k < NIN) {
                float r = rintf(W1[(size_t)n * NIN + k] / s1);
                r = fminf(fmaxf(r, -127.f), 127.f);
                q = (signed char)(int)r;
            }
            W1q[j] = q;
        } else {
            const size_t jj = j - N1;
            const int n = (int)(jj / K2), k = (int)(jj % K2);
            signed char q = 0;
            if (n < NOUT && k < NH) {
                float r = rintf(W2[(size_t)n * NH + k] / s2);
                r = fminf(fmaxf(r, -127.f), 127.f);
                q = (signed char)(int)r;
            }
            W2q[jj] = q;
        }
    }
    if (i == 0) {
        float* sc = (float*)(ws + OFF_SCALE);
        sc[0] = s1; sc[1] = s2;
    }
}

__global__ __launch_bounds__(256)
void snn_gemm1_lif(const float* __restrict__ data_t, const float* __restrict__ b1,
                   unsigned char* __restrict__ ws) {
    __shared__ __align__(16) unsigned char As[64][64];
    __shared__ __align__(16) unsigned char Bs[256][64];
    const signed char* W1q = (const signed char*)(ws + OFF_W1Q);
    unsigned char* spk1 = ws + OFF_SPK1;
    float* mem1 = (float*)(ws + OFF_MEM1);
    const float scale1 = ((const float*)(ws + OFF_SCALE))[0];

    const int tid   = threadIdx.x;
    const int lane  = tid & 31;
    const int wave  = tid >> 5;
    const int waveM = wave & 1;
    const int waveN = wave >> 1;
    const int mBase = blockIdx.x * 64;
    const int nBase = blockIdx.y * 256;
    const int half  = (lane < 16) ? 0 : 1;
    const int ln16  = lane & 15;

    v8i acc[8];
    #pragma unroll
    for (int t = 0; t < 8; ++t) acc[t] = (v8i){};

    for (int kc = 0; kc < K1_PAD / 64; ++kc) {
        #pragma unroll
        for (int g = tid; g < 1024; g += 256) {
            const int r = g >> 4, cg = g & 15;
            const int col = kc * 64 + cg * 4;
            unsigned int packed = 0u;
            if (col < NIN) {
                const float4 f = *(const float4*)(data_t + (size_t)(mBase + r) * NIN + col);
                packed = (f.x > 0.5f ? 1u : 0u)       | (f.y > 0.5f ? 1u : 0u) << 8 |
                         (f.z > 0.5f ? 1u : 0u) << 16 | (f.w > 0.5f ? 1u : 0u) << 24;
            }
            *(unsigned int*)&As[r][cg * 4] = packed;
        }
        async_copy64((unsigned)(size_t)&Bs[tid][0],
                     W1q + (size_t)(nBase + tid) * K1_PAD + kc * 64);
        if ((kc + 1) * 64 < NIN)
            __builtin_prefetch(data_t + (size_t)(mBase + (tid & 63)) * NIN + (kc + 1) * 64,
                               0, 1);
        async_wait0();
        __syncthreads();

        const int mloc = waveM * 32 + ln16;
        v8i a0, a1;
        #pragma unroll
        for (int j = 0; j < 8; ++j) {
            const int ks = ((j >> 2) << 5) + (((j >> 1) & 1) << 4) + ((j & 1) << 2) + (half << 3);
            a0[j] = *(const int*)&As[mloc][ks];
            a1[j] = *(const int*)&As[mloc + 16][ks];
        }
        const int khalf = half << 4;
        #pragma unroll
        for (int t = 0; t < 4; ++t) {
            const int nloc = waveN * 64 + t * 16 + ln16;
            v8i b;
            #pragma unroll
            for (int j = 0; j < 8; ++j) {
                const int ks = ((j >> 2) << 5) + (((j >> 1) & 1) << 4) + ((j & 1) << 2) + (khalf >> 1);
                b[j] = *(const int*)&Bs[nloc][ks];
            }
            acc[t]     = __builtin_amdgcn_wmma_i32_16x16x64_iu8(false, a0, true, b, acc[t],
                                                                false, false);
            acc[4 + t] = __builtin_amdgcn_wmma_i32_16x16x64_iu8(false, a1, true, b, acc[4 + t],
                                                                false, false);
        }
        __syncthreads();
    }

    const int rOff = (lane < 16) ? 0 : 8;
    #pragma unroll
    for (int mt = 0; mt < 2; ++mt) {
        #pragma unroll
        for (int t = 0; t < 4; ++t) {
            const int col = nBase + waveN * 64 + t * 16 + ln16;
            if (col >= NH) continue;
            const float bias = b1[col];
            #pragma unroll
            for (int r = 0; r < 8; ++r) {
                const int row = mBase + waveM * 32 + mt * 16 + r + rOff;
                const float cur = scale1 * (float)acc[mt * 4 + t][r] + bias;
                const size_t idx = (size_t)row * NH_PAD + col;
                float m = mem1[idx];
                const float reset = (m > THRV) ? THRV : 0.f;
                m = m + cur - reset;
                mem1[idx] = m;
                spk1[idx] = ((m - THRV) > 0.f) ? 1 : 0;
            }
        }
    }
}

__global__ __launch_bounds__(256)
void snn_gemm2_lif(const float* __restrict__ b2, unsigned char* __restrict__ ws,
                   float* __restrict__ out, int t) {
    __shared__ __align__(16) unsigned char Ws2[16][K2];
    const signed char* W2q = (const signed char*)(ws + OFF_W2Q);
    const unsigned char* spk1 = ws + OFF_SPK1;
    float* mem2 = (float*)(ws + OFF_MEM2);
    const float scale2 = ((const float*)(ws + OFF_SCALE))[1];

    const int tid  = threadIdx.x;
    const int lane = tid & 31;
    const int wave = tid >> 5;
    const int half = (lane < 16) ? 0 : 1;
    const int ln16 = lane & 15;
    const int mBase = blockIdx.x * 128 + wave * 16;

    {
        const int r = tid >> 4;
        const int c = (tid & 15) * 64;
        async_copy64((unsigned)(size_t)&Ws2[r][c], W2q + (size_t)r * K2 + c);
        async_wait0();
    }
    __syncthreads();

    const int m = mBase + ln16;
    v8i c = (v8i){};
    for (int kc = 0; kc < K2 / 64; ++kc) {
        v8i a, b;
        #pragma unroll
        for (int j = 0; j < 8; ++j) {
            const int ka = kc * 64 + ((j >> 2) << 5) + (((j >> 1) & 1) << 4) +
                           ((j & 1) << 2) + (half << 3);
            a[j] = *(const int*)&spk1[(size_t)m * K2 + ka];
            const int kb = kc * 64 + ((j >> 2) << 5) + (((j >> 1) & 1) << 4) + ((j & 1) << 2) + (half << 3);
            b[j] = *(const int*)&Ws2[ln16][kb];
        }
        c = __builtin_amdgcn_wmma_i32_16x16x64_iu8(false, a, true, b, c, false, false);
    }

    const int n = ln16;
    const int rOff = (lane < 16) ? 0 : 8;
    if (n < NOUT) {
        const float bias = b2[n];
        #pragma unroll
        for (int r = 0; r < 8; ++r) {
            const int row = mBase + r + rOff;
            const float cur = scale2 * (float)c[r] + bias;
            const size_t mi = (size_t)row * 16 + n;
            float mv = mem2[mi];
            const float reset = (mv > THRV) ? THRV : 0.f;
            mv = mv + cur - reset;
            mem2[mi] = mv;
            const float spk = ((mv - THRV) > 0.f) ? 1.f : 0.f;
            const size_t oidx = ((size_t)t * BATCH + row) * NOUT + n;
            out[oidx] = spk;
            out[(size_t)T_STEPS * BATCH * NOUT + oidx] = mv;
        }
    }
}

extern "C" void kernel_launch(void* const* d_in, const int* in_sizes, int n_in,
                              void* d_out, int out_size, void* d_ws, size_t ws_size,
                              hipStream_t stream) {
    const float* data = (const float*)d_in[0];
    const float* W1   = (const float*)d_in[1];
    const float* b1   = (const float*)d_in[2];
    const float* W2   = (const float*)d_in[3];
    const float* b2   = (const float*)d_in[4];
    float* out = (float*)d_out;
    unsigned char* ws = (unsigned char*)d_ws;

    snn_init  <<<512, 256, 0, stream>>>(ws);
    snn_absmax<<<512, 256, 0, stream>>>(W1, W2, ws);
    snn_quant <<<1024, 256, 0, stream>>>(W1, W2, ws);

    for (int t = 0; t < T_STEPS; ++t) {
        snn_gemm1_lif<<<dim3(BATCH / 64, NH_PAD / 256), 256, 0, stream>>>(
            data + (size_t)t * BATCH * NIN, b1, ws);
        snn_gemm2_lif<<<BATCH / 128, 256, 0, stream>>>(b2, ws, out, t);
    }
}
